// RNNEncoder_40888088658318
// MI455X (gfx1250) — matrix operand order repaired and checked by reading, not compiled, not run on hardware
//
#include <hip/hip_runtime.h>
#include <hip/hip_bf16.h>

// ---------------------------------------------------------------------------
// GRU-VAE encoder for MI455X (gfx1250, wave32, WMMA).
// 32 blocks x 16 batch rows; each block runs the whole L=512 scan with h
// resident in LDS. GEMMs use v_wmma_f32_16x16x32_f16 (f16 in, f32 accum).
// Weights converted once to row-major f16 in workspace (0.62 MB, L2-resident);
// B fragments are two 16B runs per lane (k0, k0+16), A fragments two 16B runs
// per lane from padded row-major LDS (padding kills 16-way bank conflicts).
// The ntile loop is kept rolled (#pragma unroll 1) so weight-fragment loads
// are NOT hoisted out of the time loop (previous version spilled ~600 VGPRs
// of hoisted fragments to scratch inside the hot loop).
// ---------------------------------------------------------------------------

#define SL    512   // sequence length L
#define NB    512   // batch N
#define DIN   128
#define HD    256   // hidden H
#define H3    768   // 3*H
#define DOUT  64
#define MT    16    // batch rows per block

#define XS_S  136   // xs row stride in halves (128 + 8 pad = 272B, 16B aligned)
#define H16_S 264   // h16 row stride in halves (256 + 8 pad = 528B, 16B aligned)

typedef __attribute__((ext_vector_type(16))) _Float16 v16h;
typedef __attribute__((ext_vector_type(8)))  _Float16 v8h;
typedef __attribute__((ext_vector_type(8)))  float    v8f;

// --- A fragment (16x32 f16, row M = lane&15) from row-major [16][stride] ---
// element j of v16h -> k(j) = (j>=8)*16 + (lane>>4)*8 + (j&7)
// => two contiguous 8-half (16B) chunks at k0 and k0+16.
__device__ __forceinline__ v16h frag_a(const _Float16* src, int stride, int lane, int ks) {
    int m  = lane & 15;
    int hi = lane >> 4;
    const _Float16* p = src + m * stride + ks * 32 + hi * 8;
    v8h lo = *(const v8h*)(p);
    v8h hh = *(const v8h*)(p + 16);
    return __builtin_shufflevector(lo, hh, 0,1,2,3,4,5,6,7,8,9,10,11,12,13,14,15);
}

// --- B fragment (32x16 f16, col N = lane&15) from row-major W[n][stride] ---
// B[k][n] = W[n][k]; element j -> k = (j>=8)*16 + (lane>>4)*8 + (j&7) : two 16B runs.
__device__ __forceinline__ v16h frag_b(const _Float16* W, int stride, int lane, int n0, int ks) {
    int n  = n0 + (lane & 15);
    int hi = lane >> 4;
    const _Float16* p = W + n * stride + ks * 32 + hi * 8;
    v8h lo = *(const v8h*)(p);
    v8h hh = *(const v8h*)(p + 16);
    return __builtin_shufflevector(lo, hh, 0,1,2,3,4,5,6,7,8,9,10,11,12,13,14,15);
}

__device__ __forceinline__ v8f wmma_f16(v16h a, v16h b, v8f c) {
    return __builtin_amdgcn_wmma_f32_16x16x32_f16(false, a, false, b, (short)0, c, false, false);
}

__device__ __forceinline__ float sigmoidf_(float x) {
    return 1.0f / (1.0f + __expf(-x));
}

// ---------------------------------------------------------------------------
// Weight conversion f32 -> f16 (row-major preserved)
// ---------------------------------------------------------------------------
__global__ void cvt_f32_f16_kernel(const float* __restrict__ src,
                                   _Float16* __restrict__ dst, int n) {
    int i = blockIdx.x * blockDim.x + threadIdx.x;
    if (i < n) dst[i] = (_Float16)src[i];
}

// ---------------------------------------------------------------------------
// Fused GRU scan + heads + reparameterization.
// grid = 32 blocks (batch tiles of 16), block = 256 threads (8 waves).
// ---------------------------------------------------------------------------
extern "C" __global__ void __launch_bounds__(256)
gru_vae_scan(const float* __restrict__ x,
             const float* __restrict__ b_ih,
             const float* __restrict__ b_hh,
             const float* __restrict__ b_mu,
             const float* __restrict__ b_lv,
             const float* __restrict__ eps,
             const _Float16* __restrict__ Wih,   // [768][128] f16
             const _Float16* __restrict__ Whh,   // [768][256] f16
             const _Float16* __restrict__ Wmu,   // [64][256]  f16
             const _Float16* __restrict__ Wlv,   // [64][256]  f16
             float* __restrict__ out) {
    extern __shared__ char smem[];
    // f32 segments first, f16 after; all fragment bases 16B aligned.
    float*    gxb = (float*)smem;                 // [16][768]  49152 B
    float*    ghb = gxb + MT * H3;                // [16][768]  49152 B
    float*    h32 = ghb + MT * H3;                // [16][256]  16384 B
    float*    muB = h32 + MT * HD;                // [16][64]    4096 B
    float*    lvB = muB + MT * DOUT;              // [16][64]    4096 B
    _Float16* xs  = (_Float16*)(lvB + MT * DOUT); // [16][136]   4352 B (padded)
    _Float16* h16 = xs + MT * XS_S;               // [16][264]   8448 B (padded)

    const int tid  = threadIdx.x;
    const int lane = tid & 31;
    const int wave = tid >> 5;
    const int n_base = blockIdx.x * MT;

    float* zOut  = out;
    float* muOut = out + (size_t)SL * NB * DOUT;
    float* lvOut = out + 2 * (size_t)SL * NB * DOUT;
    float* hnOut = out + 3 * (size_t)SL * NB * DOUT;

    // h0 = 0
    for (int i = tid; i < MT * HD; i += 256) {
        const int m = i >> 8, c = i & 255;
        h32[i] = 0.0f;
        h16[m * H16_S + c] = (_Float16)0.0f;
    }
    __syncthreads();

#pragma unroll 1
    for (int l = 0; l < SL; ++l) {
        // ---- stage x_t tile (16 x 128) to LDS as f16 (padded rows) ----
        const float* xt = x + ((size_t)l * NB + n_base) * DIN;
        for (int i = tid; i < MT * DIN; i += 256) {
            const int m = i >> 7, k = i & 127;
            xs[m * XS_S + k] = (_Float16)xt[i];
        }
        __syncthreads();

        // ---- GEMMs: gx = x·W_ihT + b_ih ; gh = h·W_hhT + b_hh (kept separate) ----
        v16h ax[4];
#pragma unroll
        for (int ks = 0; ks < 4; ++ks) ax[ks] = frag_a(xs, XS_S, lane, ks);
        v16h ah[8];
#pragma unroll
        for (int ks = 0; ks < 8; ++ks) ah[ks] = frag_a(h16, H16_S, lane, ks);

        // Rolled: B-fragment addresses depend on j => no LICM hoist, no spills.
#pragma unroll 1
        for (int j = 0; j < 6; ++j) {
            const int nt  = j * 8 + wave;          // 8 waves cover 128 contiguous cols
            const int col = nt * 16 + (lane & 15);
            const float bx = b_ih[col];
            const float bh = b_hh[col];
            v8f accX, accH;
#pragma unroll
            for (int r = 0; r < 8; ++r) { accX[r] = bx; accH[r] = bh; }
#pragma unroll
            for (int ks = 0; ks < 4; ++ks)
                accX = wmma_f16(ax[ks], frag_b(Wih, DIN, lane, nt * 16, ks), accX);
#pragma unroll
            for (int ks = 0; ks < 8; ++ks)
                accH = wmma_f16(ah[ks], frag_b(Whh, HD, lane, nt * 16, ks), accH);
            const int mrow = (lane >> 4) ? 8 : 0;  // D: lanes16-31 hold M=r+8
#pragma unroll
            for (int r = 0; r < 8; ++r) {
                gxb[(mrow + r) * H3 + col] = accX[r];
                ghb[(mrow + r) * H3 + col] = accH[r];
            }
        }
        __syncthreads();

        // ---- gates: thread t owns hidden column c = t across all 16 rows ----
        {
            const int c = tid;                      // 0..255 == H
#pragma unroll 4
            for (int m = 0; m < MT; ++m) {
                const float rr = sigmoidf_(gxb[m * H3 + c] + ghb[m * H3 + c]);
                const float zz = sigmoidf_(gxb[m * H3 + HD + c] + ghb[m * H3 + HD + c]);
                const float nn = tanhf(gxb[m * H3 + 2 * HD + c] + rr * ghb[m * H3 + 2 * HD + c]);
                const float hv = (1.0f - zz) * nn + zz * h32[m * HD + c];
                h32[m * HD + c] = hv;
                h16[m * H16_S + c] = (_Float16)hv;
            }
        }
        __syncthreads();

        // ---- heads: waves 0-3 -> mu tiles 0-3, waves 4-7 -> logvar tiles 0-3 ----
        {
            const _Float16* W  = (wave < 4) ? Wmu : Wlv;
            const float*    bb = (wave < 4) ? b_mu : b_lv;
            float*          ob = (wave < 4) ? muB : lvB;
            const int nt  = wave & 3;
            const int col = nt * 16 + (lane & 15);
            const float bv = bb[col];
            v8f acc;
#pragma unroll
            for (int r = 0; r < 8; ++r) acc[r] = bv;
#pragma unroll
            for (int ks = 0; ks < 8; ++ks)
                acc = wmma_f16(frag_a(h16, H16_S, lane, ks),
                               frag_b(W, HD, lane, nt * 16, ks), acc);
            const int mrow = (lane >> 4) ? 8 : 0;
#pragma unroll
            for (int r = 0; r < 8; ++r)
                ob[(mrow + r) * DOUT + col] = acc[r];
        }
        __syncthreads();

        // ---- reparameterize + write z, mu, logvar ----
        for (int i = tid; i < MT * DOUT; i += 256) {
            const int m = i >> 6, d = i & 63;
            const size_t g = ((size_t)l * NB + n_base + m) * DOUT + d;
            const float mv = muB[i];
            const float lv = lvB[i];
            zOut[g]  = mv + eps[g] * __expf(0.5f * lv);
            muOut[g] = mv;
            lvOut[g] = lv;
        }
        __syncthreads();
    }

    // ---- final hidden state h_n (1, N, H) ----
    for (int i = tid; i < MT * HD; i += 256) {
        const int m = i >> 8;       // /256
        const int hc = i & 255;
        hnOut[(size_t)(n_base + m) * HD + hc] = h32[i];
    }
}

// ---------------------------------------------------------------------------
extern "C" void kernel_launch(void* const* d_in, const int* in_sizes, int n_in,
                              void* d_out, int out_size, void* d_ws, size_t ws_size,
                              hipStream_t stream) {
    const float* x    = (const float*)d_in[0];
    const float* W_ih = (const float*)d_in[1];
    const float* b_ih = (const float*)d_in[2];
    const float* W_hh = (const float*)d_in[3];
    const float* b_hh = (const float*)d_in[4];
    const float* W_mu = (const float*)d_in[5];
    const float* b_mu = (const float*)d_in[6];
    const float* W_lv = (const float*)d_in[7];
    const float* b_lv = (const float*)d_in[8];
    const float* eps  = (const float*)d_in[9];

    // Workspace: f16 row-major weight copies (0.62 MB total).
    _Float16* Wih16 = (_Float16*)d_ws;
    _Float16* Whh16 = Wih16 + H3 * DIN;   // 98304 halves
    _Float16* Wmu16 = Whh16 + H3 * HD;    // 196608 halves
    _Float16* Wlv16 = Wmu16 + DOUT * HD;  // 16384 halves

    const int nIh = H3 * DIN, nHh = H3 * HD, nHead = DOUT * HD;
    cvt_f32_f16_kernel<<<(nIh   + 255) / 256, 256, 0, stream>>>(W_ih, Wih16, nIh);
    cvt_f32_f16_kernel<<<(nHh   + 255) / 256, 256, 0, stream>>>(W_hh, Whh16, nHh);
    cvt_f32_f16_kernel<<<(nHead + 255) / 256, 256, 0, stream>>>(W_mu, Wmu16, nHead);
    cvt_f32_f16_kernel<<<(nHead + 255) / 256, 256, 0, stream>>>(W_lv, Wlv16, nHead);

    const size_t ldsBytes =
        (size_t)(MT * H3) * 4 * 2 +    // gxb, ghb
        (size_t)(MT * HD) * 4 +        // h32
        (size_t)(MT * DOUT) * 4 * 2 +  // muB, lvB
        (size_t)(MT * XS_S) * 2 +      // xs  (f16, padded)
        (size_t)(MT * H16_S) * 2;      // h16 (f16, padded)   => 135,680 B

    gru_vae_scan<<<NB / MT, 256, ldsBytes, stream>>>(
        x, b_ih, b_hh, b_mu, b_lv, eps, Wih16, Whh16, Wmu16, Wlv16, (float*)d_out);
}
